// DeepConvWeigthNet_33741263077568
// MI455X (gfx1250) — matrix operand order repaired and checked by reading, not compiled, not run on hardware
//
#include <hip/hip_runtime.h>
#include <hip/hip_bf16.h>

typedef __attribute__((ext_vector_type(16))) _Float16 v16h;
typedef __attribute__((ext_vector_type(8)))  float    v8f;

// ---------------------------------------------------------------------------
// Weight prep: pack OIHW f32 3x3 weights into per-lane WMMA A-fragment order.
// Chunk c -> tap t = c/KCH (ky=t/3,kx=t%3), cin sub-block s = c%KCH.
// A-fragment (f16 16x16x32): lane l row m=l&15; element e (0..15):
//   K = (l>>4)*16 + e;  cin = s*32 + K
// ---------------------------------------------------------------------------
template<int CIN, int COUTP, int COUT_REAL>
__global__ __launch_bounds__(256) void prep_frag(const float* __restrict__ w,
                                                 _Float16* __restrict__ wfrag) {
  constexpr int KCH = CIN / 32;
  constexpr int CHB = COUTP / 16;
  constexpr int NC  = 9 * KCH;
  const int total = NC * CHB * 32 * 16;
  for (int i = blockIdx.x * blockDim.x + threadIdx.x; i < total;
       i += gridDim.x * blockDim.x) {
    int e    = i & 15;
    int lane = (i >> 4) & 31;
    int rest = i >> 9;
    int cb   = rest % CHB;
    int c    = rest / CHB;
    int t = c / KCH, s = c % KCH;
    int ky = t / 3, kx = t % 3;
    int m  = lane & 15;
    int hi = lane >> 4;
    int K  = hi * 16 + e;
    int cin = s * 32 + K;
    int co  = cb * 16 + m;
    float v = (co < COUT_REAL) ? w[((size_t)(co * CIN + cin) * 3 + ky) * 3 + kx] : 0.f;
    wfrag[i] = (_Float16)v;
  }
}

// ---------------------------------------------------------------------------
// conv1: 3->32 channels, 3x3 pad1, PReLU. Small K (27) -> plain VALU kernel.
// ---------------------------------------------------------------------------
__global__ __launch_bounds__(256) void conv1_prelu(
    const float* __restrict__ x, const float* __restrict__ w1,
    const float* __restrict__ b1, const float* __restrict__ a1,
    _Float16* __restrict__ out, int H, int W) {
  __shared__ float wsm[32 * 27 + 32];
  const int tid = threadIdx.x;
  for (int i = tid; i < 32 * 27; i += 256) wsm[i] = w1[i];
  for (int i = tid; i < 32; i += 256) wsm[864 + i] = b1[i];
  __syncthreads();
  const int HW = H * W;
  int gid = blockIdx.x * 256 + tid;
  int b = gid / HW, yx = gid % HW;
  int y = yx / W, xx = yx % W;
  float inv[27];
#pragma unroll
  for (int ci = 0; ci < 3; ++ci)
#pragma unroll
    for (int ky = 0; ky < 3; ++ky)
#pragma unroll
      for (int kx = 0; kx < 3; ++kx) {
        int gy = y + ky - 1, gx = xx + kx - 1;
        inv[ci * 9 + ky * 3 + kx] =
            (gy >= 0 && gy < H && gx >= 0 && gx < W)
                ? x[((size_t)(b * 3 + ci) * H + gy) * W + gx] : 0.f;
      }
  const float al = *a1;
  for (int co = 0; co < 32; ++co) {
    float s = wsm[864 + co];
    const float* wv = &wsm[co * 27];
#pragma unroll
    for (int k = 0; k < 27; ++k) s = fmaf(wv[k], inv[k], s);
    s = s > 0.f ? s : al * s;
    out[(size_t)(b * 32 + co) * HW + yx] = (_Float16)s;
  }
}

// ---------------------------------------------------------------------------
// Implicit-GEMM 3x3 conv via v_wmma_f32_16x16x32_f16.
// 256 threads = 8 waves; wave (cb,pw) computes a 16(outch)x16(px) tile.
//  - Packed A (weights) slab copied global->LDS with GLOBAL_LOAD_ASYNC_TO_LDS
//    (ASYNCcnt, drained via s_wait_asynccnt) -- contiguous, VGPR-free path.
//  - Input tile [3 rows][PX+2 cols][CIN ch] staged in LDS (cin contiguous ->
//    B-fragment = one aligned 32B ds read); interior copied with aligned
//    b128 loads, only the 2 halo columns scalar.
// ---------------------------------------------------------------------------
template<int CIN, int COUTP, int COUT_REAL, bool ACT>
__global__ __launch_bounds__(256) void conv3x3_wmma(
    const _Float16* __restrict__ in, const _Float16* __restrict__ wfrag,
    const float* __restrict__ bias, const float* __restrict__ alpha,
    _Float16* __restrict__ out, int H, int W) {
  constexpr int CHB = COUTP / 16;   // out-channel blocks
  constexpr int PXB = 8 / CHB;      // pixel blocks per WG
  constexpr int PX  = PXB * 16;     // pixels per WG (one row segment)
  constexpr int PXT = PX + 2;       // + halo
  constexpr int KCH = CIN / 32;     // K sub-chunks per tap
  constexpr int NC  = 9 * KCH;      // total 32-wide K chunks
  constexpr int WHALF = NC * CHB * 32 * 16;   // packed weight halves
  constexpr int WCHUNK = WHALF * 2 / 16;      // 16B chunks of weight slab

  __shared__ __align__(32) _Float16 smem[3 * PXT * CIN];
  __shared__ __align__(32) _Float16 wsm[WHALF];

  const int tid  = threadIdx.x;
  const int lane = tid & 31;
  const int wave = tid >> 5;
  const int cb   = wave % CHB;
  const int pw   = wave / CHB;

  const int px0 = blockIdx.x * PX;
  const int y   = blockIdx.y;
  const int b   = blockIdx.z;

  // --- async copy of packed weights into LDS (16B per lane per issue) ------
  {
    const unsigned lds0 = (unsigned)(uintptr_t)&wsm[0];  // LDS aperture: low 32b
    const unsigned long long g0 = (unsigned long long)(uintptr_t)wfrag;
    for (int i = tid; i < WCHUNK; i += 256) {
      unsigned ldsa = lds0 + (unsigned)(i * 16);
      unsigned long long ga = g0 + (unsigned long long)(i * 16);
      asm volatile("global_load_async_to_lds_b128 %0, %1, off"
                   :: "v"(ldsa), "v"(ga) : "memory");
    }
  }

  // --- stage input tile: interior as aligned b128, halo columns scalar -----
  {
    constexpr int NGV = PX / 8;               // 8-half vector groups per row
    for (int i = tid; i < 3 * CIN * NGV; i += 256) {
      int g   = i % NGV;
      int rest = i / NGV;
      int ci  = rest % CIN;
      int row = rest / CIN;
      int gy  = y + row - 1;
      int gx0 = px0 + g * 8;                  // (..)*2B is 16B aligned
      union { uint4 u; _Float16 h[8]; } vv;
      if (gy >= 0 && gy < H)
        vv.u = *(const uint4*)(in + ((size_t)(b * CIN + ci) * H + gy) * W + gx0);
      else
        vv.u = make_uint4(0, 0, 0, 0);
      const int colb = 1 + g * 8;
#pragma unroll
      for (int j = 0; j < 8; ++j)
        smem[(row * PXT + colb + j) * CIN + ci] = vv.h[j];
    }
    for (int i = tid; i < 3 * CIN * 2; i += 256) {  // halo: col 0 and PXT-1
      int side = i & 1;
      int rest = i >> 1;
      int ci  = rest % CIN;
      int row = rest / CIN;
      int gy  = y + row - 1;
      int col = side ? (PXT - 1) : 0;
      int gx  = px0 + col - 1;
      _Float16 v = (_Float16)0.f;
      if (gy >= 0 && gy < H && gx >= 0 && gx < W)
        v = in[((size_t)(b * CIN + ci) * H + gy) * W + gx];
      smem[(row * PXT + col) * CIN + ci] = v;
    }
  }

  asm volatile("s_wait_asynccnt 0x0" ::: "memory");   // weights landed in LDS
  __syncthreads();

  const int n  = lane & 15;
  const int hi = lane >> 4;

  v8f acc = {};
#pragma unroll
  for (int c = 0; c < NC; ++c) {
    const int t = c / KCH, s = c % KCH;
    const int ky = t / 3, kx = t % 3;
    v16h a = *(const v16h*)(&wsm[((c * CHB + cb) * 32 + lane) * 16]);
    const int col = pw * 16 + n + kx;
    v16h bf = *(const v16h*)(&smem[(ky * PXT + col) * CIN + s * 32 + hi * 16]);
    acc = __builtin_amdgcn_wmma_f32_16x16x32_f16(
        false, a, false, bf, (short)0, acc, false, false);
  }

  const float al = ACT ? *alpha : 0.f;
  const int px = px0 + pw * 16 + n;
#pragma unroll
  for (int r = 0; r < 8; ++r) {
    const int m  = r + (hi << 3);
    const int co = cb * 16 + m;
    if (co < COUT_REAL) {
      float v = acc[r] + bias[co];
      if (ACT) v = v > 0.f ? v : al * v;
      out[((size_t)(b * COUT_REAL + co) * H + y) * W + px] = (_Float16)v;
    }
  }
}

// ---------------------------------------------------------------------------
// Per-(head,b,ch) global mean of y (for channel attention). 60 blocks.
// ---------------------------------------------------------------------------
__global__ __launch_bounds__(256) void mean60(const _Float16* __restrict__ y,
                                              float* __restrict__ means, int HW) {
  __shared__ float red[256];
  const int id = blockIdx.x;
  const _Float16* p = y + (size_t)id * HW;
  float s = 0.f;
  for (int i = threadIdx.x; i < HW; i += 256) s += (float)p[i];
  red[threadIdx.x] = s;
  __syncthreads();
  for (int st = 128; st > 0; st >>= 1) {
    if (threadIdx.x < st) red[threadIdx.x] += red[threadIdx.x + st];
    __syncthreads();
  }
  if (threadIdx.x == 0) means[id] = red[0] / (float)HW;
}

// z = sigmoid(W2 @ relu(W1 @ mean)), per (head, batch, out-channel). 60 lanes.
__global__ void ca_z(const float* __restrict__ means,
                     const float* w1a, const float* w2a,
                     const float* w1b, const float* w2b,
                     const float* w1c, const float* w2c,
                     float* __restrict__ zs) {
  int id = threadIdx.x;
  if (id >= 60) return;
  int head = id / 20, rem = id % 20, b = rem / 5, o = rem % 5;
  const float* W1 = head == 0 ? w1a : head == 1 ? w1b : w1c;
  const float* W2 = head == 0 ? w2a : head == 1 ? w2b : w2c;
  const float* m = means + (head * 4 + b) * 5;
  float acc = 0.f;
  for (int i = 0; i < 5; ++i) {
    float t = 0.f;
    for (int j = 0; j < 5; ++j) t = fmaf(W1[i * 5 + j], m[j], t);
    t = t > 0.f ? t : 0.f;
    acc = fmaf(W2[o * 5 + i], t, acc);
  }
  zs[id] = 1.f / (1.f + __expf(-acc));
}

// h = softmax_ch(y * z), per pixel, all 3 heads in one grid.
__global__ __launch_bounds__(256) void head_softmax(
    const _Float16* __restrict__ y, const float* __restrict__ zs,
    _Float16* __restrict__ h, int HW) {
  int gid = blockIdx.x * 256 + threadIdx.x;  // over 3*B*HW
  int p  = gid % HW;
  int hb = gid / HW;                          // head*B + b
  float v[5];
  float mx = -1e30f;
#pragma unroll
  for (int c = 0; c < 5; ++c) {
    v[c] = (float)y[((size_t)hb * 5 + c) * HW + p] * zs[hb * 5 + c];
    mx = fmaxf(mx, v[c]);
  }
  float s = 0.f;
#pragma unroll
  for (int c = 0; c < 5; ++c) { v[c] = __expf(v[c] - mx); s += v[c]; }
  float inv = 1.f / s;
#pragma unroll
  for (int c = 0; c < 5; ++c)
    h[((size_t)hb * 5 + c) * HW + p] = (_Float16)(v[c] * inv);
}

// Stage, pass 1: horizontal window sums for box radii 2/7/12 (separable box).
__global__ __launch_bounds__(256) void stage_hbox(const float* __restrict__ u,
                                                  float* __restrict__ th,
                                                  int H, int W, int BC) {
  int gid = blockIdx.x * 256 + threadIdx.x;  // over BC*HW
  int HW = H * W;
  int yx = gid % HW, bc = gid / HW;
  int y = yx / W, x = yx % W;
  const float* row = u + (size_t)bc * HW + (size_t)y * W;
  float s25 = 0.f, s15 = 0.f, s5 = 0.f;
#pragma unroll
  for (int dx = -12; dx <= 12; ++dx) {
    int gx = x + dx;
    float v = (gx >= 0 && gx < W) ? row[gx] : 0.f;
    s25 += v;
    if (dx >= -7 && dx <= 7) s15 += v;
    if (dx >= -2 && dx <= 2) s5 += v;
  }
  size_t o = (size_t)bc * HW + yx;
  size_t S = (size_t)BC * HW;
  th[o] = s5; th[S + o] = s15; th[2 * S + o] = s25;
}

// Stage, pass 2: vertical sums, sharpen+identity taps, blend with h weights.
__global__ __launch_bounds__(256) void stage_vcombine(
    const float* __restrict__ u, const float* __restrict__ th,
    const _Float16* __restrict__ h, float* __restrict__ out,
    int H, int W, int B) {
  int gid = blockIdx.x * 256 + threadIdx.x;  // over B*3*HW
  int HW = H * W;
  int yx = gid % HW, bc = gid / HW;
  int b = bc / 3;
  int y = yx / W, x = yx % W;
  size_t S = (size_t)B * 3 * HW;
  const float* t5  = th + (size_t)bc * HW;
  const float* t15 = th + S + (size_t)bc * HW;
  const float* t25 = th + 2 * S + (size_t)bc * HW;
  float v5 = 0.f, v15 = 0.f, v25 = 0.f;
#pragma unroll
  for (int dy = -12; dy <= 12; ++dy) {
    int gy = y + dy;
    if (gy < 0 || gy >= H) continue;
    size_t ro = (size_t)gy * W + x;
    v25 += t25[ro];
    if (dy >= -7 && dy <= 7) v15 += t15[ro];
    if (dy >= -2 && dy <= 2) v5  += t5[ro];
  }
  const float* up = u + (size_t)bc * HW;
  float c0 = up[yx];
  float um = (y > 0)     ? up[yx - W] : 0.f;
  float ud = (y < H - 1) ? up[yx + W] : 0.f;
  float ul = (x > 0)     ? up[yx - 1] : 0.f;
  float ur = (x < W - 1) ? up[yx + 1] : 0.f;
  float sharp = 5.f * c0 - um - ud - ul - ur;
  float hw_[5];
#pragma unroll
  for (int c = 0; c < 5; ++c)
    hw_[c] = (float)h[(size_t)(b * 5 + c) * HW + yx];
  float o = hw_[0] * sharp + hw_[1] * c0 + hw_[2] * v5 * (1.f / 25.f) +
            hw_[3] * v15 * (1.f / 225.f) + hw_[4] * v25 * (1.f / 625.f);
  out[(size_t)bc * HW + yx] = o;
}

// ---------------------------------------------------------------------------
extern "C" void kernel_launch(void* const* d_in, const int* in_sizes, int n_in,
                              void* d_out, int out_size, void* d_ws, size_t ws_size,
                              hipStream_t stream) {
  (void)in_sizes; (void)n_in; (void)out_size; (void)ws_size;
  const int B = 4, H = 512, W = 512, HW = H * W;

  const float* x  = (const float*)d_in[0];
  const float* w1 = (const float*)d_in[1];
  const float* b1 = (const float*)d_in[2];
  const float* a1 = (const float*)d_in[3];
  const float* w2 = (const float*)d_in[4];
  const float* b2 = (const float*)d_in[5];
  const float* a2 = (const float*)d_in[6];
  const float* w3 = (const float*)d_in[7];
  const float* b3 = (const float*)d_in[8];
  const float* a3 = (const float*)d_in[9];
  const float* hw_[3] = {(const float*)d_in[10], (const float*)d_in[14], (const float*)d_in[18]};
  const float* hb_[3] = {(const float*)d_in[11], (const float*)d_in[15], (const float*)d_in[19]};
  const float* cw1[3] = {(const float*)d_in[12], (const float*)d_in[16], (const float*)d_in[20]};
  const float* cw2[3] = {(const float*)d_in[13], (const float*)d_in[17], (const float*)d_in[21]};

  char* ws = (char*)d_ws;
  auto al256 = [](size_t v) { return (v + 255) & ~(size_t)255; };
  size_t off = 0;
  // bodyA: conv1 out (32ch f16); later reused for conv3 out (32ch f16)
  _Float16* bodyA = (_Float16*)(ws + off); off = al256(off + (size_t)B * 32 * HW * 2);
  // bodyB: conv2 out (64ch f16); region reused after conv3 for y/h/th/means/zs
  size_t bodyB_off = off;
  _Float16* bodyB = (_Float16*)(ws + off); off = al256(off + (size_t)B * 64 * HW * 2);
  size_t r = bodyB_off;
  _Float16* y_all = (_Float16*)(ws + r); r = al256(r + (size_t)3 * B * 5 * HW * 2);
  _Float16* h_all = (_Float16*)(ws + r); r = al256(r + (size_t)3 * B * 5 * HW * 2);
  float* th    = (float*)(ws + r); r = al256(r + (size_t)3 * B * 3 * HW * 4);
  float* means = (float*)(ws + r); r = al256(r + 256);
  float* zs    = (float*)(ws + r); r = al256(r + 256);  // all fits inside bodyB
  float* u1 = (float*)(ws + off); off = al256(off + (size_t)B * 3 * HW * 4);
  float* u2 = (float*)(ws + off); off = al256(off + (size_t)B * 3 * HW * 4);
  _Float16* wf2 = (_Float16*)(ws + off); off = al256(off + 18432 * 2);
  _Float16* wf3 = (_Float16*)(ws + off); off = al256(off + 18432 * 2);
  _Float16* wfh[3];
  for (int i = 0; i < 3; ++i) { wfh[i] = (_Float16*)(ws + off); off = al256(off + 4608 * 2); }

  // pack weights into WMMA A-fragment order
  prep_frag<32, 64, 64><<<72, 256, 0, stream>>>(w2, wf2);
  prep_frag<64, 32, 32><<<72, 256, 0, stream>>>(w3, wf3);
  for (int i = 0; i < 3; ++i)
    prep_frag<32, 16, 5><<<18, 256, 0, stream>>>(hw_[i], wfh[i]);

  // body convs
  conv1_prelu<<<B * HW / 256, 256, 0, stream>>>(x, w1, b1, a1, bodyA, H, W);
  conv3x3_wmma<32, 64, 64, true>
      <<<dim3(W / 32, H, B), 256, 0, stream>>>(bodyA, wf2, b2, a2, bodyB, H, W);
  conv3x3_wmma<64, 32, 32, true>
      <<<dim3(W / 64, H, B), 256, 0, stream>>>(bodyB, wf3, b3, a3, bodyA, H, W);

  // heads: y = conv(body), Cout padded 5->16
  for (int i = 0; i < 3; ++i)
    conv3x3_wmma<32, 16, 5, false>
        <<<dim3(W / 128, H, B), 256, 0, stream>>>(bodyA, wfh[i], hb_[i], hb_[i],
                                                  y_all + (size_t)i * B * 5 * HW, H, W);
  mean60<<<60, 256, 0, stream>>>(y_all, means, HW);
  ca_z<<<1, 64, 0, stream>>>(means, cw1[0], cw2[0], cw1[1], cw2[1], cw1[2], cw2[2], zs);
  head_softmax<<<3 * B * HW / 256, 256, 0, stream>>>(y_all, zs, h_all, HW);

  // stages: separable boxes + sharpen + identity, blended with h weights
  const float* us[3] = {x, u1, u2};
  float* os[3] = {u1, u2, (float*)d_out};
  for (int s = 0; s < 3; ++s) {
    stage_hbox<<<B * 3 * HW / 256, 256, 0, stream>>>(us[s], th, H, W, B * 3);
    stage_vcombine<<<B * 3 * HW / 256, 256, 0, stream>>>(
        us[s], th, h_all + (size_t)s * B * 5 * HW, os[s], H, W, B);
  }
}
